// HybridFeatureExtractor_52776558133916
// MI455X (gfx1250) — matrix operand order repaired and checked by reading, not compiled, not run on hardware
//
#include <hip/hip_runtime.h>
#include <hip/hip_bf16.h>

#define NN  16384
#define EE  524288
#define KNB 16      // k for kNN
#define HID 64

typedef __attribute__((ext_vector_type(16))) _Float16 v16h;
typedef __attribute__((ext_vector_type(8)))  float    v8f;

__device__ __forceinline__ v8f wmma16(v16h a, v16h b, v8f c) {
  // D = A(16x32 f16) * B(32x16 f16) + C(16x16 f32)
  return __builtin_amdgcn_wmma_f32_16x16x32_f16(
      false, a, false, b, (short)0, c, false, false);
}

// A-operand fetch: q points at row base (+kbase). Lane layout per ISA 7.12.2:
// k = 16*(vg/4) + 8*hi + 2*(vg&3) (+0/+1), pairs contiguous.
__device__ __forceinline__ v16h lds_load_a(const _Float16* q, int hi) {
  v16h a;
#pragma unroll
  for (int vg = 0; vg < 8; ++vg) {
    int k = ((vg >> 2) << 4) + (hi << 3) + ((vg & 3) << 1);
    a[2 * vg]     = q[k];
    a[2 * vg + 1] = q[k + 1];
  }
  return a;
}

// B-operand fetch from K-contiguous (transposed) storage: q points at col base
// (+kbase). k = 16*(vg/4) + 8*hi + 2*(vg&3) (+0/+1).
__device__ __forceinline__ v16h lds_load_b(const _Float16* q, int hi) {
  v16h b;
#pragma unroll
  for (int vg = 0; vg < 8; ++vg) {
    int k = ((vg >> 2) << 4) + (hi << 3) + ((vg & 3) << 1);
    b[2 * vg]     = q[k];
    b[2 * vg + 1] = q[k + 1];
  }
  return b;
}

// Async global -> LDS copy (16B per lane), ASYNCcnt-tracked (CDNA5).
// VDST holds the LDS byte address (low 32 bits of the generic shared pointer),
// VADDR the 64-bit global address (GV mode, saddr = off).
__device__ __forceinline__ void async_copy_b128(void* lds_dst, const void* gsrc) {
  unsigned lds_off = (unsigned)(size_t)lds_dst;
  asm volatile("global_load_async_to_lds_b128 %0, %1, off"
               :: "v"(lds_off), "v"(gsrc)
               : "memory");
}
__device__ __forceinline__ void wait_async_le1() {
  asm volatile("s_wait_asynccnt 0x1" ::: "memory");
}
__device__ __forceinline__ void wait_async_le0() {
  asm volatile("s_wait_asynccnt 0x0" ::: "memory");
}

// ---------------------------------------------------------------------------
// Kernel 1: static EdgeConv. m = relu(relu([xi, xj-xi] W1 + b1) W2 + b2),
// scatter-max into x1 (uint atomicMax on non-negative floats; x1 pre-zeroed).
// ---------------------------------------------------------------------------
__global__ void __launch_bounds__(128) edgeconv1_kernel(
    const float* __restrict__ x, const int* __restrict__ ei,
    const float* __restrict__ W1, const float* __restrict__ b1,
    const float* __restrict__ W2, const float* __restrict__ b2,
    float* __restrict__ x1) {
  __shared__ _Float16 sW1t[64 * 32];      // [n][k], k padded 6->32
  __shared__ _Float16 sW2t[64 * 64];      // [n][k]
  __shared__ float sB1[64], sB2[64];
  __shared__ _Float16 sFeat[4][16 * 32];  // per-wave edge features
  __shared__ _Float16 sH[4][16 * 64];     // per-wave hidden
  __shared__ int sDst[4][16];

  const int tid = threadIdx.x;
  for (int i = tid; i < 64 * 32; i += 128) {
    int n = i >> 5, k = i & 31;
    sW1t[i] = (k < 6) ? (_Float16)W1[k * 64 + n] : (_Float16)0.0f;
  }
  for (int i = tid; i < 64 * 64; i += 128) {
    int n = i >> 6, k = i & 63;
    sW2t[i] = (_Float16)W2[k * 64 + n];
  }
  if (tid < 64) { sB1[tid] = b1[tid]; sB2[tid] = b2[tid]; }
  __syncthreads();

  const int wave = tid >> 5, lane = tid & 31;
  const int lo = lane & 15, hi = lane >> 4;
  _Float16* feat = sFeat[wave];
  _Float16* hbuf = sH[wave];

  const int slot = blockIdx.x * 4 + wave;
  const int nslots = gridDim.x * 4;
  for (int tile = slot; tile < EE / 16; tile += nslots) {
    if (hi == 0) {  // stage 16 edges (lanes 0..15)
      int e = tile * 16 + lo;
      int s = ei[e];           // edge_index[0] = src j
      int d = ei[EE + e];      // edge_index[1] = dst i
      sDst[wave][lo] = d;
      _Float16* fr = feat + lo * 32;
#pragma unroll
      for (int c = 0; c < 3; ++c) {
        float xd = x[d * 3 + c];
        float xs = x[s * 3 + c];
        fr[c]     = (_Float16)xd;
        fr[3 + c] = (_Float16)(xs - xd);
      }
#pragma unroll
      for (int c = 6; c < 32; ++c) fr[c] = (_Float16)0.0f;
    }
    // layer 1: [16x32(pad)] @ [32x64]
#pragma unroll
    for (int nt = 0; nt < 4; ++nt) {
      v8f acc = {};
      v16h a = lds_load_a(feat + lo * 32, hi);
      v16h b = lds_load_b(sW1t + (nt * 16 + lo) * 32, hi);
      acc = wmma16(a, b, acc);
      int c = nt * 16 + lo;
      float bias = sB1[c];
#pragma unroll
      for (int v = 0; v < 8; ++v) {
        int r = v + 8 * hi;
        float hv = acc[v] + bias;
        hbuf[r * 64 + c] = (_Float16)(hv > 0.0f ? hv : 0.0f);
      }
    }
    // layer 2: [16x64] @ [64x64] + relu + scatter-max
#pragma unroll
    for (int nt = 0; nt < 4; ++nt) {
      v8f acc = {};
#pragma unroll
      for (int kt = 0; kt < 2; ++kt) {
        v16h a = lds_load_a(hbuf + lo * 64 + kt * 32, hi);
        v16h b = lds_load_b(sW2t + (nt * 16 + lo) * 64 + kt * 32, hi);
        acc = wmma16(a, b, acc);
      }
      int c = nt * 16 + lo;
      float bias = sB2[c];
#pragma unroll
      for (int v = 0; v < 8; ++v) {
        int r = v + 8 * hi;
        float m = acc[v] + bias;
        m = m > 0.0f ? m : 0.0f;
        atomicMax((unsigned int*)&x1[(size_t)sDst[wave][r] * 64 + c],
                  __float_as_uint(m));
      }
    }
  }
}

// ---------------------------------------------------------------------------
// Kernel 2: x1 -> f16 copy + squared norms of the f16-rounded rows.
// ---------------------------------------------------------------------------
__global__ void x1_half_sq_kernel(const float* __restrict__ x1,
                                  _Float16* __restrict__ x1h,
                                  float* __restrict__ sq) {
  int r = blockIdx.x * blockDim.x + threadIdx.x;
  if (r >= NN) return;
  const float4* p = (const float4*)(x1 + (size_t)r * 64);
  _Float16* q = x1h + (size_t)r * 64;
  float acc = 0.0f;
#pragma unroll
  for (int c4 = 0; c4 < 16; ++c4) {
    float4 v = p[c4];
    _Float16 h0 = (_Float16)v.x, h1 = (_Float16)v.y;
    _Float16 h2 = (_Float16)v.z, h3 = (_Float16)v.w;
    q[c4 * 4 + 0] = h0; q[c4 * 4 + 1] = h1;
    q[c4 * 4 + 2] = h2; q[c4 * 4 + 3] = h3;
    acc += (float)h0 * (float)h0 + (float)h1 * (float)h1 +
           (float)h2 * (float)h2 + (float)h3 * (float)h3;
  }
  sq[r] = acc;
}

// ---------------------------------------------------------------------------
// Kernel 3: fused kNN. 64 query rows/block, scan all 16384 candidates in
// 64-col tiles via WMMA Gram subtiles. Candidate tiles are double-buffered
// and staged with GLOBAL_LOAD_ASYNC_TO_LDS_B128 (ASYNCcnt), overlapping the
// copy of tile t+1 with WMMA + top-k work on tile t.
// ---------------------------------------------------------------------------
#define KNN_SMEM 74240
__global__ void __launch_bounds__(512) knn_kernel(
    const _Float16* __restrict__ x1h, const float* __restrict__ sq,
    int* __restrict__ knn) {
  __shared__ __align__(16) char smem[KNN_SMEM];
  _Float16* sQ   = (_Float16*)smem;              // 8192 B, live whole kernel
  float*    sQsq = (float*)(smem + 8192);        // 256 B
  _Float16* sC0  = (_Float16*)(smem + 8448);     // 8192 B  ping
  _Float16* sC1  = (_Float16*)(smem + 16640);    // 8192 B  pong
  float*    sCsq = (float*)(smem + 24832);       // 256 B
  float*    sDist= (float*)(smem + 25088);       // 16384 B -> 41472
  float*    mD   = (float*)(smem + 8448);        // 32768 B (merge phase)
  int*      mI   = (int*)(smem + 41472);         // 32768 B (merge phase)
  _Float16* sC[2] = { sC0, sC1 };

  const int tid = threadIdx.x;
  const int qbase = blockIdx.x * 64;
  const int wave = tid >> 5, lane = tid & 31;
  const int rt = wave >> 2, ct = wave & 3;
  const int lo = lane & 15, hi = lane >> 4;

  // stage queries: 64 rows x 64 halves = 512 int4 chunks, one per thread
  ((int4*)sQ)[tid] = ((const int4*)(x1h + (size_t)qbase * 64))[tid];
  if (tid < 64) sQsq[tid] = sq[qbase + tid];

  // kick off async copy of candidate tile 0 (8 KB, 16 B per thread)
  async_copy_b128((char*)sC0 + tid * 16, (const char*)x1h + tid * 16);
  __syncthreads();

  float best[16]; int bidx[16];
#pragma unroll
  for (int i = 0; i < 16; ++i) { best[i] = 3.0e38f; bidx[i] = 0; }
  float curMax = 3.0e38f;

  const int row = tid >> 3, slotc = tid & 7;

  for (int ctile = 0; ctile < NN / 64; ++ctile) {
    const int cbase = ctile * 64;
    const int cur = ctile & 1;
    if (ctile + 1 < NN / 64) {  // prefetch next tile into the other buffer
      async_copy_b128((char*)sC[cur ^ 1] + tid * 16,
                      (const char*)x1h + (size_t)(cbase + 64) * 128 + tid * 16);
      wait_async_le1();         // in-order completion => tile `ctile` resident
    } else {
      wait_async_le0();
    }
    if (tid < 64) sCsq[tid] = sq[cbase + tid];
    __syncthreads();

    // 16 waves -> 4x4 grid of 16x16 dot-product subtiles (K=64 -> 2 WMMA)
    {
      const _Float16* sCc = sC[cur];
      v8f acc = {};
#pragma unroll
      for (int kt = 0; kt < 2; ++kt) {
        v16h a = lds_load_a(sQ + (rt * 16 + lo) * 64 + kt * 32, hi);
        v16h b = lds_load_b(sCc + (ct * 16 + lo) * 64 + kt * 32, hi);
        acc = wmma16(a, b, acc);
      }
      int c = ct * 16 + lo;
      float csq = sCsq[c];
#pragma unroll
      for (int v = 0; v < 8; ++v) {
        int r = rt * 16 + v + 8 * hi;
        sDist[r * 64 + c] = sQsq[r] + csq - 2.0f * acc[v];
      }
    }
    __syncthreads();

    // top-k update: 8 threads per row, 8 candidates each
#pragma unroll
    for (int t = 0; t < 8; ++t) {
      int c = slotc + t * 8;
      float d = sDist[row * 64 + c];
      if (d < curMax) {
        float mv = best[0]; int rm = 0;
#pragma unroll
        for (int i = 1; i < 16; ++i)
          if (best[i] > mv) { mv = best[i]; rm = i; }
#pragma unroll
        for (int i = 0; i < 16; ++i)
          if (i == rm) { best[i] = d; bidx[i] = cbase + c; }
        curMax = best[0];
#pragma unroll
        for (int i = 1; i < 16; ++i) curMax = fmaxf(curMax, best[i]);
      }
    }
    __syncthreads();
  }

  // merge: dump 8 partial top-16 lists per row, then select 16 smallest
  {
    float* pd = mD + row * 128 + slotc * 16;
    int*   pi = mI + row * 128 + slotc * 16;
#pragma unroll
    for (int i = 0; i < 16; ++i) { pd[i] = best[i]; pi[i] = bidx[i]; }
  }
  __syncthreads();
  if (tid < 64) {
    float* pd = mD + tid * 128;
    int*   pi = mI + tid * 128;
    int q = qbase + tid;
    for (int j = 0; j < 16; ++j) {
      float bv = pd[0]; int bi = 0;
      for (int t = 1; t < 128; ++t)
        if (pd[t] < bv) { bv = pd[t]; bi = t; }
      knn[q * 16 + j] = pi[bi];
      pd[bi] = 3.0e38f;
    }
  }
}

// ---------------------------------------------------------------------------
// Kernel 4: DynamicEdgeConv. Each 16-edge tile == one dst node, so the
// segment-max is an in-register column max; plain stores, no atomics.
// ---------------------------------------------------------------------------
__global__ void __launch_bounds__(128) edgeconv2_kernel(
    const float* __restrict__ x1, const int* __restrict__ knn,
    const float* __restrict__ W3, const float* __restrict__ b3,
    const float* __restrict__ W4, const float* __restrict__ b4,
    float* __restrict__ out) {
  __shared__ _Float16 sW3t[64 * 128];      // [n][k]
  __shared__ _Float16 sW4t[64 * 64];       // [n][k]
  __shared__ float sB3[64], sB4[64];
  __shared__ _Float16 sFeat[4][16 * 128];  // per-wave edge features
  __shared__ _Float16 sH[4][16 * 64];

  const int tid = threadIdx.x;
  for (int i = tid; i < 64 * 128; i += 128) {
    int n = i >> 7, k = i & 127;
    sW3t[i] = (_Float16)W3[k * 64 + n];
  }
  for (int i = tid; i < 64 * 64; i += 128) {
    int n = i >> 6, k = i & 63;
    sW4t[i] = (_Float16)W4[k * 64 + n];
  }
  if (tid < 64) { sB3[tid] = b3[tid]; sB4[tid] = b4[tid]; }
  __syncthreads();

  const int wave = tid >> 5, lane = tid & 31;
  const int lo = lane & 15, hi = lane >> 4;
  _Float16* feat = sFeat[wave];
  _Float16* hbuf = sH[wave];

  const int slot = blockIdx.x * 4 + wave;
  const int nslots = gridDim.x * 4;
  for (int node = slot; node < NN; node += nslots) {
    const float* xi = x1 + (size_t)node * 64;
    if (hi == 0) {  // feat[t][0:64] = x_i (same for all 16 rows)
#pragma unroll
      for (int c4 = 0; c4 < 16; ++c4) {
        float4 v = ((const float4*)xi)[c4];
        _Float16* fr = feat + lo * 128 + c4 * 4;
        fr[0] = (_Float16)v.x; fr[1] = (_Float16)v.y;
        fr[2] = (_Float16)v.z; fr[3] = (_Float16)v.w;
      }
    } else {        // feat[t][64:128] = x_j - x_i
      int s = knn[node * KNB + lo];
      const float* xj = x1 + (size_t)s * 64;
#pragma unroll
      for (int c4 = 0; c4 < 16; ++c4) {
        float4 vj = ((const float4*)xj)[c4];
        float4 vi = ((const float4*)xi)[c4];
        _Float16* fr = feat + lo * 128 + 64 + c4 * 4;
        fr[0] = (_Float16)(vj.x - vi.x); fr[1] = (_Float16)(vj.y - vi.y);
        fr[2] = (_Float16)(vj.z - vi.z); fr[3] = (_Float16)(vj.w - vi.w);
      }
    }
    // layer 1: [16x128] @ [128x64]
#pragma unroll
    for (int nt = 0; nt < 4; ++nt) {
      v8f acc = {};
#pragma unroll
      for (int kt = 0; kt < 4; ++kt) {
        v16h a = lds_load_a(feat + lo * 128 + kt * 32, hi);
        v16h b = lds_load_b(sW3t + (nt * 16 + lo) * 128 + kt * 32, hi);
        acc = wmma16(a, b, acc);
      }
      int c = nt * 16 + lo;
      float bias = sB3[c];
#pragma unroll
      for (int v = 0; v < 8; ++v) {
        int r = v + 8 * hi;
        float hv = acc[v] + bias;
        hbuf[r * 64 + c] = (_Float16)(hv > 0.0f ? hv : 0.0f);
      }
    }
    // layer 2: [16x64] @ [64x64] + relu + column max -> out[node]
#pragma unroll
    for (int nt = 0; nt < 4; ++nt) {
      v8f acc = {};
#pragma unroll
      for (int kt = 0; kt < 2; ++kt) {
        v16h a = lds_load_a(hbuf + lo * 64 + kt * 32, hi);
        v16h b = lds_load_b(sW4t + (nt * 16 + lo) * 64 + kt * 32, hi);
        acc = wmma16(a, b, acc);
      }
      int c = nt * 16 + lo;
      float bias = sB4[c];
      float colmax = 0.0f;  // post-relu values are >= 0
#pragma unroll
      for (int v = 0; v < 8; ++v) {
        float m = acc[v] + bias;
        m = m > 0.0f ? m : 0.0f;
        colmax = fmaxf(colmax, m);
      }
      colmax = fmaxf(colmax, __shfl_xor(colmax, 16));
      if (hi == 0) out[(size_t)node * 64 + c] = colmax;
    }
  }
}

// ---------------------------------------------------------------------------
extern "C" void kernel_launch(void* const* d_in, const int* in_sizes, int n_in,
                              void* d_out, int out_size, void* d_ws,
                              size_t ws_size, hipStream_t stream) {
  (void)in_sizes; (void)n_in; (void)out_size; (void)ws_size;
  const float* x  = (const float*)d_in[0];
  const int*   ei = (const int*)d_in[1];    // [2,E] int32 (JAX x64 off)
  const float* W1 = (const float*)d_in[3];
  const float* b1 = (const float*)d_in[4];
  const float* W2 = (const float*)d_in[5];
  const float* b2 = (const float*)d_in[6];
  const float* W3 = (const float*)d_in[7];
  const float* b3 = (const float*)d_in[8];
  const float* W4 = (const float*)d_in[9];
  const float* b4 = (const float*)d_in[10];

  char* ws = (char*)d_ws;
  size_t off = 0;
  float* x1 = (float*)(ws + off);        off += (size_t)NN * 64 * 4;  // 4 MB
  _Float16* x1h = (_Float16*)(ws + off); off += (size_t)NN * 64 * 2;  // 2 MB
  float* sqv = (float*)(ws + off);       off += (size_t)NN * 4;       // 64 KB
  int* knn = (int*)(ws + off);           off += (size_t)NN * KNB * 4; // 1 MB

  hipMemsetAsync(x1, 0, (size_t)NN * 64 * 4, stream);
  edgeconv1_kernel<<<2048, 128, 0, stream>>>(x, ei, W1, b1, W2, b2, x1);
  x1_half_sq_kernel<<<NN / 256, 256, 0, stream>>>(x1, x1h, sqv);
  knn_kernel<<<NN / 64, 512, 0, stream>>>(x1h, sqv, knn);
  edgeconv2_kernel<<<512, 128, 0, stream>>>(x1, knn, W3, b3, W4, b4,
                                            (float*)d_out);
}
